// LayoutLMv3Encoder_49873160241681
// MI455X (gfx1250) — matrix operand order repaired and checked by reading, not compiled, not run on hardware
//
#include <hip/hip_runtime.h>

// ---------------------------------------------------------------------------
// LayoutLMv3 encoder, CDNA5 (gfx1250) WMMA bf16 implementation, v3.
//   L=12  B=4  S=512  D=768  H=12  DH=64  FF=3072
// - All GEMMs + attention use v_wmma_f32_16x16x32_bf16 (fp32 accumulate).
// - Weights pre-packed per layer into LDS-ready bf16 k-pair layout.
// - Double-buffered LDS staging via global_load_async_to_lds_b128:
//   async loads for tile k+1 overlap the 8 WMMAs of tile k.
// ---------------------------------------------------------------------------

typedef __attribute__((ext_vector_type(16))) __bf16 v16bf;
typedef __attribute__((ext_vector_type(8)))  float  v8f;

#define LNUM 12
#define BB   4
#define SS   512
#define DD   768
#define HH   12
#define DHH  64
#define FFD  3072
#define MM   (BB * SS)          // 2048
#define BHN  (BB * HH)          // 48

#if defined(__AMDGCN__) && __has_builtin(__builtin_amdgcn_global_load_async_to_lds_b128) && __has_builtin(__builtin_amdgcn_s_wait_asynccnt)
#define USE_ASYNC_LDS 1
#else
#define USE_ASYNC_LDS 0
#endif

static __device__ __forceinline__ unsigned short f2bf_bits(float x) {
  __bf16 b = (__bf16)x;
  return __builtin_bit_cast(unsigned short, b);
}
static __device__ __forceinline__ unsigned packbf(float lo, float hi) {
  return (unsigned)f2bf_bits(lo) | ((unsigned)f2bf_bits(hi) << 16);
}
static __device__ __forceinline__ v8f wmma_bf16(v16bf a, v16bf b, v8f c) {
  return __builtin_amdgcn_wmma_f32_16x16x32_bf16(
      /*neg_a=*/false, a, /*neg_b=*/false, b,
      /*c_mod=*/(short)0, c, /*reuse_a=*/false, /*reuse_b=*/false);
}

#if USE_ASYNC_LDS
// Matches the builtin's parameter type (per compiler diagnostic):
// "__attribute__((__vector_size__(4 * sizeof(int)))) int *"
typedef int vint4_ls __attribute__((__vector_size__(16)));
static __device__ __forceinline__ void async_ld16(const void* g, void* l) {
  __builtin_amdgcn_global_load_async_to_lds_b128((vint4_ls*)g, (vint4_ls*)l,
                                                 0, 0);
}
#endif

union FragBF { v16bf v; unsigned u[8]; };

// ---------------------------------------------------------------------------
// bias[bh, i, j] = (rel + rel2) * (1/sqrt(64)) + mask[b, j]
// ---------------------------------------------------------------------------
__global__ void bias_prep_kernel(const float* __restrict__ rel,
                                 const float* __restrict__ rel2,
                                 const float* __restrict__ mask,
                                 float* __restrict__ out) {
  size_t i = (size_t)blockIdx.x * blockDim.x + threadIdx.x;
  size_t stride = (size_t)gridDim.x * blockDim.x;
  const size_t total = (size_t)BHN * SS * SS;
  for (; i < total; i += stride) {
    int b = (int)(i / ((size_t)HH * SS * SS));
    int j = (int)(i & (SS - 1));
    out[i] = (rel[i] + rel2[i]) * 0.125f + mask[b * SS + j];
  }
}

// fp32 -> (fp32 copy, bf16 copy)
__global__ void cvt_kernel(const float* __restrict__ x,
                           float* __restrict__ outF,
                           unsigned short* __restrict__ outB) {
  size_t i = (size_t)blockIdx.x * blockDim.x + threadIdx.x;
  size_t stride = (size_t)gridDim.x * blockDim.x;
  const size_t total = (size_t)MM * DD;
  for (; i < total; i += stride) {
    float v = x[i];
    outF[i] = v;
    outB[i] = f2bf_bits(v);
  }
}

// ---------------------------------------------------------------------------
// Weight pre-pack: W fp32 [K][N] -> Wp dword [N][K/2], dword = bf16 pair
// (k, k+1) at column n.  This is exactly the B-fragment LDS layout.
// ---------------------------------------------------------------------------
__global__ void pack_w_kernel(const float* __restrict__ W,
                              unsigned* __restrict__ Wp, int K, int N) {
  const int kd2 = K >> 1;
  size_t i = (size_t)blockIdx.x * blockDim.x + threadIdx.x;
  size_t stride = (size_t)gridDim.x * blockDim.x;
  const size_t total = (size_t)N * kd2;
  for (; i < total; i += stride) {
    int n = (int)(i / kd2);
    int kdw = (int)(i - (size_t)n * kd2);
    Wp[i] = packbf(W[(size_t)(2 * kdw) * N + n],
                   W[(size_t)(2 * kdw + 1) * N + n]);
  }
}

// ---------------------------------------------------------------------------
// GEMM: C[M,N] = A(bf16,[M,K]) * W + bias[N]   (+ epilogue)
// W pre-packed as Wp[N][K/2] dwords.
// Block: 256 threads = 8 waves (4x2), tile 128(M) x 128(N), K-step 32.
// Each wave computes a 32x64 tile (2x4 WMMA C-frags) -> 8 WMMA / K-step.
// Double-buffered LDS: async loads for tile k+1 issued before computing k.
// mode 0: scatter bf16 -> Q  [bh][s][d]
// mode 1: scatter bf16 -> KT [bh][d/2][s] (d-pair packed dwords)
// mode 2: scatter bf16 -> VP [bh][s/2][d] (s-pair packed dwords)
// mode 3: fp32 out = acc + bias + resid      (pre-layernorm buffer)
// mode 4: bf16 out = gelu(acc + bias)        (FFN intermediate)
// ---------------------------------------------------------------------------
__global__ __launch_bounds__(256) void gemm_bf16_kernel(
    const unsigned short* __restrict__ A, const unsigned* __restrict__ Wp,
    const float* __restrict__ bias, const float* __restrict__ resid,
    float* __restrict__ outF, unsigned short* __restrict__ outBF,
    int M, int N, int K, int mode) {
  // Row stride 20 dwords: keeps every 16B async destination aligned and
  // column reads across 16 rows conflict-free (gcd(20,64) spread).
  __shared__ unsigned Asu[2][128][20];   // [buf][m][k-pair] 128x32 bf16 tile
  __shared__ unsigned Bsu[2][128][20];   // [buf][n][k-pair] 32x128 bf16 tile

  const int tid  = threadIdx.x;
  const int lane = tid & 31;
  const int wid  = tid >> 5;
  const int wm   = wid & 3;            // 4 wave-rows
  const int wn   = wid >> 2;           // 2 wave-cols
  const int half = lane >> 4;
  const int l16  = lane & 15;

  const int m0 = blockIdx.x * 128;
  const int n0 = blockIdx.y * 128;
  const int kd2 = K >> 1;

  const v8f vz = {0.f, 0.f, 0.f, 0.f, 0.f, 0.f, 0.f, 0.f};
  v8f c[2][4];
#pragma unroll
  for (int i = 0; i < 2; ++i)
#pragma unroll
    for (int j = 0; j < 4; ++j) c[i][j] = vz;

  FragBF afr[2], bfr[4];

  const int rT = tid >> 1;             // 0..127 : tile row
  const int cT = tid & 1;              // 0..1   : 8-dword half-row

  // Stage one 32-K tile into LDS buffer `buf`.
  auto stage = [&](int kt, int buf) {
    const int k0 = kt << 5;
    const unsigned short* gA = A + (size_t)(m0 + rT) * K + k0 + cT * 16;
    const unsigned* gB = Wp + (size_t)(n0 + rT) * kd2 + (k0 >> 1) + cT * 8;
#if USE_ASYNC_LDS
    async_ld16(gA,     &Asu[buf][rT][cT * 8]);
    async_ld16(gA + 8, &Asu[buf][rT][cT * 8 + 4]);
    async_ld16(gB,     &Bsu[buf][rT][cT * 8]);
    async_ld16(gB + 4, &Bsu[buf][rT][cT * 8 + 4]);
#else
    const uint4 a0 = *(const uint4*)gA;
    const uint4 a1 = *(const uint4*)(gA + 8);
    const uint4 b0 = *(const uint4*)gB;
    const uint4 b1 = *(const uint4*)(gB + 4);
    Asu[buf][rT][cT * 8 + 0] = a0.x; Asu[buf][rT][cT * 8 + 1] = a0.y;
    Asu[buf][rT][cT * 8 + 2] = a0.z; Asu[buf][rT][cT * 8 + 3] = a0.w;
    Asu[buf][rT][cT * 8 + 4] = a1.x; Asu[buf][rT][cT * 8 + 5] = a1.y;
    Asu[buf][rT][cT * 8 + 6] = a1.z; Asu[buf][rT][cT * 8 + 7] = a1.w;
    Bsu[buf][rT][cT * 8 + 0] = b0.x; Bsu[buf][rT][cT * 8 + 1] = b0.y;
    Bsu[buf][rT][cT * 8 + 2] = b0.z; Bsu[buf][rT][cT * 8 + 3] = b0.w;
    Bsu[buf][rT][cT * 8 + 4] = b1.x; Bsu[buf][rT][cT * 8 + 5] = b1.y;
    Bsu[buf][rT][cT * 8 + 6] = b1.z; Bsu[buf][rT][cT * 8 + 7] = b1.w;
#endif
  };

  const int nkt = K >> 5;
  stage(0, 0);
#if USE_ASYNC_LDS
  __builtin_amdgcn_s_wait_asynccnt(0);
#endif
  __syncthreads();

  for (int kt = 0; kt < nkt; ++kt) {
    const int cur = kt & 1;
    // Prefetch next tile into the alternate buffer; overlaps the WMMAs below.
    if (kt + 1 < nkt) stage(kt + 1, cur ^ 1);

    // ---- fragment loads from LDS (current buffer) ----
#pragma unroll
    for (int i = 0; i < 2; ++i) {
      int mr = wm * 32 + i * 16 + l16;
#pragma unroll
      for (int v = 0; v < 8; ++v)
        afr[i].u[v] = Asu[cur][mr][half * 8 + v];
    }
#pragma unroll
    for (int j = 0; j < 4; ++j) {
      int nc = wn * 64 + j * 16 + l16;
#pragma unroll
      for (int v = 0; v < 8; ++v)
        bfr[j].u[v] = Bsu[cur][nc][half * 8 + v];
    }
#pragma unroll
    for (int i = 0; i < 2; ++i)
#pragma unroll
      for (int j = 0; j < 4; ++j)
        c[i][j] = wmma_bf16(afr[i].v, bfr[j].v, c[i][j]);

    // Drain this wave's async loads, then block-sync before the buffer swap.
#if USE_ASYNC_LDS
    __builtin_amdgcn_s_wait_asynccnt(0);
#endif
    __syncthreads();
  }

  // ---- epilogue ----
#pragma unroll
  for (int i = 0; i < 2; ++i) {
#pragma unroll
    for (int j = 0; j < 4; ++j) {
      int n = n0 + wn * 64 + j * 16 + l16;
      float bval = bias[n];
#pragma unroll
      for (int r = 0; r < 8; ++r) {
        int m = m0 + wm * 32 + i * 16 + r + half * 8;
        float val = c[i][j][r] + bval;
        if (mode == 3) {
          val += resid[(size_t)m * N + n];
          outF[(size_t)m * N + n] = val;
        } else if (mode == 4) {
          float gx = 0.5f * val * (1.0f + erff(val * 0.70710678118f));
          outBF[(size_t)m * N + n] = f2bf_bits(gx);
        } else {
          int b = m >> 9, s = m & (SS - 1);
          int h = n >> 6, d = n & (DHH - 1);
          int bh = b * HH + h;
          size_t off;
          if (mode == 0)
            off = ((size_t)bh * SS + s) * DHH + d;                         // Q
          else if (mode == 1)
            off = (((size_t)bh * 32 + (d >> 1)) * SS + s) * 2 + (d & 1);   // KT
          else
            off = (((size_t)bh * 256 + (s >> 1)) * DHH + d) * 2 + (s & 1); // VP
          outBF[off] = f2bf_bits(val);
        }
      }
    }
  }
}

// ---------------------------------------------------------------------------
// Fused attention: per block = one (b,h) pair, 64 q rows; 4 waves, each wave
// owns 16 q rows; flash-style online softmax over 32-key tiles.
// Q : bf16 [bh][s][64] ; KT: dword [bh][d/2][s] ; VP: dword [bh][s/2][d]
// ---------------------------------------------------------------------------
__global__ __launch_bounds__(128) void attn_kernel(
    const unsigned short* __restrict__ Q, const unsigned* __restrict__ KT,
    const unsigned* __restrict__ VP, const float* __restrict__ BIAS,
    unsigned short* __restrict__ CTX) {
  __shared__ unsigned short Ps[4][16][36];  // per-wave probs tile 16x32 (+pad)

  const int tid  = threadIdx.x;
  const int lane = tid & 31;
  const int wid  = tid >> 5;
  const int half = lane >> 4;
  const int l16  = lane & 15;
  const int bh   = blockIdx.y;
  const int b    = bh / HH;
  const int h    = bh % HH;
  const int qW   = blockIdx.x * 64 + wid * 16;

  FragBF qa[2], kb[2][2], pa, vb[4];
  const v8f vz = {0.f, 0.f, 0.f, 0.f, 0.f, 0.f, 0.f, 0.f};
  v8f sc[2], c[4];
#pragma unroll
  for (int j = 0; j < 4; ++j) c[j] = vz;

  const unsigned short* qrow = Q + ((size_t)bh * SS + qW + l16) * DHH;
#pragma unroll
  for (int fi = 0; fi < 2; ++fi)
#pragma unroll
    for (int v = 0; v < 8; ++v) {
      int d = fi * 32 + half * 16 + v * 2;
      qa[fi].u[v] = *(const unsigned*)(qrow + d);
    }

  float rmax[8], rsum[8];
#pragma unroll
  for (int r = 0; r < 8; ++r) { rmax[r] = -1e30f; rsum[r] = 0.f; }

  const float scale = 0.125f;  // 1/sqrt(64)
  const float* biasRow = BIAS + ((size_t)bh * SS + qW) * SS;

  for (int kt = 0; kt < 16; ++kt) {
    const int kb0 = kt * 32;
#pragma unroll
    for (int ds = 0; ds < 2; ++ds)
#pragma unroll
      for (int nf = 0; nf < 2; ++nf) {
        int kcol = kb0 + nf * 16 + l16;
#pragma unroll
        for (int v = 0; v < 8; ++v) {
          int dp = ds * 16 + half * 8 + v;
          kb[ds][nf].u[v] = KT[((size_t)bh * 32 + dp) * SS + kcol];
        }
      }
    sc[0] = vz; sc[1] = vz;
#pragma unroll
    for (int ds = 0; ds < 2; ++ds) {
      sc[0] = wmma_bf16(qa[ds].v, kb[ds][0].v, sc[0]);
      sc[1] = wmma_bf16(qa[ds].v, kb[ds][1].v, sc[1]);
    }

    float s0[8], s1[8], rm[8];
#pragma unroll
    for (int r = 0; r < 8; ++r) {
      int qr = r + half * 8;
      const float* bp = biasRow + (size_t)qr * SS + kb0;
      s0[r] = sc[0][r] * scale + bp[l16];
      s1[r] = sc[1][r] * scale + bp[16 + l16];
      rm[r] = fmaxf(s0[r], s1[r]);
    }
#pragma unroll
    for (int r = 0; r < 8; ++r)
#pragma unroll
      for (int off = 1; off < 16; off <<= 1)
        rm[r] = fmaxf(rm[r], __shfl_xor(rm[r], off, 16));
    float f[8];
#pragma unroll
    for (int r = 0; r < 8; ++r) {
      float nm = fmaxf(rmax[r], rm[r]);
      f[r] = __expf(rmax[r] - nm);
      rmax[r] = nm;
      s0[r] = __expf(s0[r] - nm);
      s1[r] = __expf(s1[r] - nm);
      float rs = s0[r] + s1[r];
#pragma unroll
      for (int off = 1; off < 16; off <<= 1) rs += __shfl_xor(rs, off, 16);
      rsum[r] = rsum[r] * f[r] + rs;
    }
#pragma unroll
    for (int j = 0; j < 4; ++j)
#pragma unroll
      for (int r = 0; r < 8; ++r) c[j][r] *= f[r];

#pragma unroll
    for (int r = 0; r < 8; ++r) {
      int row = r + half * 8;
      Ps[wid][row][l16]      = f2bf_bits(s0[r]);
      Ps[wid][row][16 + l16] = f2bf_bits(s1[r]);
    }
    __builtin_amdgcn_wave_barrier();
#pragma unroll
    for (int v = 0; v < 8; ++v) {
      int kk = half * 16 + v * 2;
      pa.u[v] = *(const unsigned*)&Ps[wid][l16][kk];
    }

#pragma unroll
    for (int j = 0; j < 4; ++j)
#pragma unroll
      for (int v = 0; v < 8; ++v) {
        int kp = (kb0 >> 1) + half * 8 + v;
        vb[j].u[v] = VP[((size_t)bh * 256 + kp) * DHH + j * 16 + l16];
      }
#pragma unroll
    for (int j = 0; j < 4; ++j) c[j] = wmma_bf16(pa.v, vb[j].v, c[j]);
  }

#pragma unroll
  for (int r = 0; r < 8; ++r) {
    float inv = 1.0f / rsum[r];
    int q = qW + r + half * 8;
#pragma unroll
    for (int j = 0; j < 4; ++j) {
      float val = c[j][r] * inv;
      CTX[((size_t)(b * SS + q)) * DD + h * DHH + j * 16 + l16] = f2bf_bits(val);
    }
  }
}

// ---------------------------------------------------------------------------
// LayerNorm over D=768; one block per row; writes fp32 + bf16 copies.
// ---------------------------------------------------------------------------
__global__ __launch_bounds__(256) void ln_kernel(
    const float* __restrict__ X, const float* __restrict__ G,
    const float* __restrict__ Bt, float* __restrict__ outF,
    unsigned short* __restrict__ outB) {
  __shared__ float red[2][8];
  const int row = blockIdx.x;
  const int tid = threadIdx.x;
  const float* x = X + (size_t)row * DD;
  float v0 = x[tid], v1 = x[tid + 256], v2 = x[tid + 512];
  float s = v0 + v1 + v2;
  float q = v0 * v0 + v1 * v1 + v2 * v2;
#pragma unroll
  for (int off = 1; off < 32; off <<= 1) {
    s += __shfl_xor(s, off, 32);
    q += __shfl_xor(q, off, 32);
  }
  int wid = tid >> 5, lane = tid & 31;
  if (lane == 0) { red[0][wid] = s; red[1][wid] = q; }
  __syncthreads();
  float ts = 0.f, tq = 0.f;
#pragma unroll
  for (int i = 0; i < 8; ++i) { ts += red[0][i]; tq += red[1][i]; }
  const float mean = ts * (1.0f / DD);
  const float var  = tq * (1.0f / DD) - mean * mean;
  const float inv  = rsqrtf(var + 1e-5f);
  float* of = outF + (size_t)row * DD;
  unsigned short* ob = outB + (size_t)row * DD;
#pragma unroll
  for (int k = 0; k < 3; ++k) {
    int col = tid + 256 * k;
    float y = (x[col] - mean) * inv * G[col] + Bt[col];
    of[col] = y;
    ob[col] = f2bf_bits(y);
  }
}

// ---------------------------------------------------------------------------
// Host-side orchestration
// ---------------------------------------------------------------------------
extern "C" void kernel_launch(void* const* d_in, const int* in_sizes, int n_in,
                              void* d_out, int out_size, void* d_ws,
                              size_t ws_size, hipStream_t stream) {
  const float* hs   = (const float*)d_in[0];
  const float* mask = (const float*)d_in[1];
  const float* rel  = (const float*)d_in[2];
  const float* rel2 = (const float*)d_in[3];
  const float* Wq = (const float*)d_in[4];  const float* bq = (const float*)d_in[5];
  const float* Wk = (const float*)d_in[6];  const float* bk = (const float*)d_in[7];
  const float* Wv = (const float*)d_in[8];  const float* bv = (const float*)d_in[9];
  const float* Wo = (const float*)d_in[10]; const float* bo = (const float*)d_in[11];
  const float* g1 = (const float*)d_in[12]; const float* b1 = (const float*)d_in[13];
  const float* Wi = (const float*)d_in[14]; const float* bi = (const float*)d_in[15];
  const float* Wo2= (const float*)d_in[16]; const float* bo2= (const float*)d_in[17];
  const float* g2 = (const float*)d_in[18]; const float* b2 = (const float*)d_in[19];

  char* ws = (char*)d_ws;
  const size_t szBias = (size_t)BHN * SS * SS * 4;   // 50,331,648
  const size_t szF    = (size_t)MM * DD * 4;         //  6,291,456
  const size_t szB    = (size_t)MM * DD * 2;         //  3,145,728
  const size_t szHead = (size_t)BHN * SS * DHH * 2;  //  3,145,728
  const size_t szInt  = (size_t)MM * FFD * 2;        // 12,582,912
  const size_t szWdd  = (size_t)DD * (DD / 2) * 4;   //  1,179,648
  const size_t szWdf  = (size_t)FFD * (DD / 2) * 4;  //  4,718,592 (and D*FF/2)

  size_t off = 0;
  auto take = [&](size_t bytes) {
    char* p = ws + off;
    off += (bytes + 255) & ~(size_t)255;
    return p;
  };
  float*          biasT = (float*)take(szBias);
  float*          hF    = (float*)take(szF);
  unsigned short* hB    = (unsigned short*)take(szB);
  unsigned short* qB    = (unsigned short*)take(szHead);
  unsigned*       kT    = (unsigned*)take(szHead);
  unsigned*       vP    = (unsigned*)take(szHead);
  unsigned short* ctxB  = (unsigned short*)take(szB);
  float*          tmpF  = (float*)take(szF);
  float*          attF  = (float*)take(szF);
  unsigned short* attB  = (unsigned short*)take(szB);
  unsigned short* intB  = (unsigned short*)take(szInt);
  unsigned*       wqP   = (unsigned*)take(szWdd);
  unsigned*       wkP   = (unsigned*)take(szWdd);
  unsigned*       wvP   = (unsigned*)take(szWdd);
  unsigned*       woP   = (unsigned*)take(szWdd);
  unsigned*       wiP   = (unsigned*)take(szWdf);
  unsigned*       wo2P  = (unsigned*)take(szWdf);

  bias_prep_kernel<<<dim3(4096), 256, 0, stream>>>(rel, rel2, mask, biasT);
  cvt_kernel<<<dim3(3072), 256, 0, stream>>>(hs, hF, hB);

  const dim3 gD(MM / 128, DD / 128);    // (16, 6)
  const dim3 gFF(MM / 128, FFD / 128);  // (16, 24)
  const dim3 gAtt(SS / 64, BHN);        // (8, 48)

  for (int l = 0; l < LNUM; ++l) {
    const float* wq  = Wq  + (size_t)l * DD * DD;
    const float* wk  = Wk  + (size_t)l * DD * DD;
    const float* wv  = Wv  + (size_t)l * DD * DD;
    const float* wo  = Wo  + (size_t)l * DD * DD;
    const float* wi  = Wi  + (size_t)l * DD * FFD;
    const float* wo2 = Wo2 + (size_t)l * FFD * DD;

    pack_w_kernel<<<dim3(1152), 256, 0, stream>>>(wq, wqP, DD, DD);
    pack_w_kernel<<<dim3(1152), 256, 0, stream>>>(wk, wkP, DD, DD);
    pack_w_kernel<<<dim3(1152), 256, 0, stream>>>(wv, wvP, DD, DD);
    pack_w_kernel<<<dim3(1152), 256, 0, stream>>>(wo, woP, DD, DD);
    pack_w_kernel<<<dim3(2304), 256, 0, stream>>>(wi, wiP, DD, FFD);
    pack_w_kernel<<<dim3(2304), 256, 0, stream>>>(wo2, wo2P, FFD, DD);

    gemm_bf16_kernel<<<gD, 256, 0, stream>>>(hB, wqP, bq + l * DD, nullptr,
                                             nullptr, qB, MM, DD, DD, 0);
    gemm_bf16_kernel<<<gD, 256, 0, stream>>>(hB, wkP, bk + l * DD, nullptr,
                                             nullptr, (unsigned short*)kT,
                                             MM, DD, DD, 1);
    gemm_bf16_kernel<<<gD, 256, 0, stream>>>(hB, wvP, bv + l * DD, nullptr,
                                             nullptr, (unsigned short*)vP,
                                             MM, DD, DD, 2);
    attn_kernel<<<gAtt, 128, 0, stream>>>(qB, kT, vP, biasT, ctxB);
    gemm_bf16_kernel<<<gD, 256, 0, stream>>>(ctxB, woP, bo + l * DD, hF,
                                             tmpF, nullptr, MM, DD, DD, 3);
    ln_kernel<<<dim3(MM), 256, 0, stream>>>(tmpF, g1 + l * DD, b1 + l * DD,
                                            attF, attB);
    gemm_bf16_kernel<<<gFF, 256, 0, stream>>>(attB, wiP, bi + l * FFD, nullptr,
                                              nullptr, intB, MM, FFD, DD, 4);
    gemm_bf16_kernel<<<gD, 256, 0, stream>>>(intB, wo2P, bo2 + l * DD, attF,
                                             tmpF, nullptr, MM, DD, FFD, 3);
    ln_kernel<<<dim3(MM), 256, 0, stream>>>(tmpF, g2 + l * DD, b2 + l * DD,
                                            hF, hB);
  }

  (void)hipMemcpyAsync(d_out, hF, szF, hipMemcpyDeviceToDevice, stream);
}
